// EmbedInit_18098992185556
// MI455X (gfx1250) — matrix operand order repaired and checked by reading, not compiled, not run on hardware
//
#include <hip/hip_runtime.h>
#include <hip/hip_bf16.h>

typedef __attribute__((ext_vector_type(16))) _Float16 v16h;
typedef __attribute__((ext_vector_type(8)))  _Float16 v8h;
typedef __attribute__((ext_vector_type(8)))  float    v8f;

#define IN_DIM  64
#define HID_DIM 128
#define OUT_DIM 64
#define BN_EPS  1e-5f

// ---------------------------------------------------------------- degree / dinv
__global__ void deg_init_kernel(float* __restrict__ deg, int n) {
  int i = blockIdx.x * blockDim.x + threadIdx.x;
  if (i < n) deg[i] = 1.0f;  // self-loop contributes 1 to every node's degree
}

__global__ void deg_edges_kernel(const long long* __restrict__ ei,
                                 float* __restrict__ deg, int E) {
  int e = blockIdx.x * blockDim.x + threadIdx.x;
  if (e < E) {
    long long d = ei[(long long)E + e];
    unsafeAtomicAdd(&deg[d], 1.0f);
  }
}

__global__ void deg_to_dinv_kernel(float* __restrict__ deg, int n) {
  int i = blockIdx.x * blockDim.x + threadIdx.x;
  if (i < n) deg[i] = __frsqrt_rn(deg[i]);  // deg >= 1 always
}

// ---------------------------------------------------------------- f32 -> f16
__global__ void cvt_f16_kernel(const float* __restrict__ src,
                               _Float16* __restrict__ dst, long long n) {
  long long i = (long long)blockIdx.x * blockDim.x + threadIdx.x;
  if (i < n) dst[i] = (_Float16)src[i];
}

// W [K x NC] f32 row-major  ->  Wt [NC x K] f16 row-major (B operand, K-contig per column)
__global__ void w_transpose_f16_kernel(const float* __restrict__ W,
                                       _Float16* __restrict__ Wt, int K, int NC) {
  int idx = blockIdx.x * blockDim.x + threadIdx.x;
  if (idx >= K * NC) return;
  int n = idx / K, k = idx % K;
  Wt[idx] = (_Float16)W[k * NC + n];
}

// ---------------------------------------------------------------- WMMA GEMM
// C[M x NC] (f32) = A[M x K] (f16, row-major) * B, with B given as Bt[NC x K] (f16).
// K, NC compile-time so K-loop/N-loop fully unroll:
//   - Bt (16 KB) staged once per workgroup into LDS, B fragments fed by ds_load_b128
//   - A fragments hoisted: loaded once per wave, reused across all NC/16 column tiles
// Fragment layouts per CDNA5 ISA 7.12.2:
//   A (16x32 f16): lane = 16*h + m. elems 0..7 = K[kc + 8h .. +7], elems 8..15 = K[kc+16+8h .. +7]
//   B (32x16 f16): lane = 16*h + n. elems 0..7 = K[kc + 8h .. +7], elems 8..15 = K[kc+16+8h .. +7]
//   C/D (16x16 f32): vgpr r, lane = 16*h + n -> element (m = r + 8h, n)
template <int K, int NC>
__global__ void wmma_gemm_f16_kernel(const _Float16* __restrict__ A,
                                     const _Float16* __restrict__ Bt,
                                     float* __restrict__ C, int M) {
  constexpr int KC = K / 32;           // 32-wide K chunks per WMMA
  __shared__ alignas(16) _Float16 Bs[NC * K];

  // cooperative LDS fill: 8 halves (16B) per thread per iteration
  for (int i = threadIdx.x * 8; i < NC * K; i += blockDim.x * 8)
    *(v8h*)(Bs + i) = *(const v8h*)(Bt + i);
  __syncthreads();

  int wave = (blockIdx.x * blockDim.x + threadIdx.x) >> 5;
  int lane = threadIdx.x & 31;
  int m0 = wave * 16;
  if (m0 >= M) return;                 // wave-uniform: EXEC stays all-ones for WMMA
  int lrow  = lane & 15;
  int lhalf = lane >> 4;

  // hoist all A fragments for this 16-row strip (K=128 -> 32 VGPRs)
  const _Float16* arow = A + (size_t)(m0 + lrow) * K;
  union { v16h v; v8h h[2]; } a[KC];
#pragma unroll
  for (int kc = 0; kc < KC; ++kc) {
    a[kc].h[0] = *(const v8h*)(arow + kc * 32 + lhalf * 8);
    a[kc].h[1] = *(const v8h*)(arow + kc * 32 + 16 + lhalf * 8);
  }

#pragma unroll
  for (int n0 = 0; n0 < NC; n0 += 16) {
    const _Float16* brow = Bs + (n0 + lrow) * K;
    v8f acc = {};
#pragma unroll
    for (int kc = 0; kc < KC; ++kc) {
      union { v16h v; v8h h[2]; } b;
      b.h[0] = *(const v8h*)(brow + kc * 32 + lhalf * 8);       // ds_load_b128
      b.h[1] = *(const v8h*)(brow + kc * 32 + 16 + lhalf * 8);  // ds_load_b128
      acc = __builtin_amdgcn_wmma_f32_16x16x32_f16(false, a[kc].v, false, b.v,
                                                   (short)0, acc, false, false);
    }
    float* ccol = C + (size_t)(m0 + lhalf * 8) * NC + n0 + lrow;
#pragma unroll
    for (int r = 0; r < 8; ++r) ccol[(size_t)r * NC] = acc[r];
  }
}

// ---------------------------------------------------------------- aggregation
// agg = xw * dinv[row]^2 + bias   (self-loop term folded in, bias folded in)
__global__ void init_agg_kernel(const float* __restrict__ xw,
                                const float* __restrict__ dinv,
                                const float* __restrict__ bias,
                                float* __restrict__ agg, int n, int D) {
  long long idx = (long long)blockIdx.x * blockDim.x + threadIdx.x;
  if (idx >= (long long)n * D) return;
  int row = (int)(idx / D);
  int c   = (int)(idx % D);
  float di = dinv[row];
  agg[idx] = xw[idx] * di * di + bias[c];
}

// Scatter-add over edges: D/4 lanes per edge, float4 gather, 4x f32 atomics to L2.
__global__ void edge_agg_kernel(const long long* __restrict__ ei,
                                const float* __restrict__ dinv,
                                const float* __restrict__ xw,
                                float* __restrict__ agg, int E, int D) {
  int qpe = D >> 2;                    // float4 quads per edge
  long long t = (long long)blockIdx.x * blockDim.x + threadIdx.x;
  if (t >= (long long)E * qpe) return;
  int e = (int)(t / qpe);
  int q = (int)(t % qpe);
  long long s = ei[e];
  long long d = ei[(long long)E + e];
  float norm = dinv[s] * dinv[d];
  const float4 v = *(const float4*)(xw + (size_t)s * D + q * 4);
  float* dst = agg + (size_t)d * D + q * 4;
  unsafeAtomicAdd(dst + 0, v.x * norm);
  unsafeAtomicAdd(dst + 1, v.y * norm);
  unsafeAtomicAdd(dst + 2, v.z * norm);
  unsafeAtomicAdd(dst + 3, v.w * norm);
}

// ---------------------------------------------------------------- batch norm
__global__ void bn_zero_kernel(float* __restrict__ stats) {
  if (threadIdx.x < 128) stats[threadIdx.x] = 0.0f;
}

__global__ void bn_reduce_kernel(const float* __restrict__ x,
                                 float* __restrict__ stats, int n) {
  int c      = threadIdx.x & 63;
  int rloc   = threadIdx.x >> 6;       // 4 rows in flight per block
  int rstep  = (blockDim.x >> 6);
  float s = 0.0f, ss = 0.0f;
  for (long long row = (long long)blockIdx.x * rstep + rloc; row < n;
       row += (long long)gridDim.x * rstep) {
    float v = x[row * OUT_DIM + c];
    s  += v;
    ss += v * v;
  }
  unsafeAtomicAdd(&stats[c], s);
  unsafeAtomicAdd(&stats[64 + c], ss);
}

__global__ void bn_finalize_kernel(float* __restrict__ stats,
                                   const float* __restrict__ gamma,
                                   const float* __restrict__ beta, int n) {
  int c = threadIdx.x;
  if (c < OUT_DIM) {
    float inv_n = 1.0f / (float)n;
    float mean  = stats[c] * inv_n;
    float var   = stats[64 + c] * inv_n - mean * mean;   // biased variance
    float sc    = gamma[c] * __frsqrt_rn(var + BN_EPS);
    stats[128 + c] = sc;
    stats[192 + c] = beta[c] - mean * sc;
  }
}

__global__ void bn_apply_kernel(const float* __restrict__ x,
                                const float* __restrict__ stats,
                                float* __restrict__ out, int n) {
  long long idx = (long long)blockIdx.x * blockDim.x + threadIdx.x;
  if (idx >= (long long)n * OUT_DIM) return;
  int c = (int)(idx % OUT_DIM);
  out[idx] = x[idx] * stats[128 + c] + stats[192 + c];
}

// ---------------------------------------------------------------- launcher
extern "C" void kernel_launch(void* const* d_in, const int* in_sizes, int n_in,
                              void* d_out, int out_size, void* d_ws, size_t ws_size,
                              hipStream_t stream) {
  const float*     embed = (const float*)d_in[0];
  const long long* ei    = (const long long*)d_in[1];   // int64 per reference
  const float*     W1    = (const float*)d_in[2];
  const float*     b1    = (const float*)d_in[3];
  const float*     W2    = (const float*)d_in[4];
  const float*     b2    = (const float*)d_in[5];
  const float*     gamma = (const float*)d_in[6];
  const float*     beta  = (const float*)d_in[7];
  float*           out   = (float*)d_out;

  const int N = in_sizes[0] / IN_DIM;      // 100000 (multiple of 16)
  const int E = in_sizes[1] / 2;           // 1600000

  // Workspace carve-out (256B-aligned). xw2 aliases xw1, agg2 aliases h.
  size_t off = 0;
  auto carve = [&](size_t bytes) -> char* {
    char* p = (char*)d_ws + off;
    off += (bytes + 255) & ~(size_t)255;
    return p;
  };
  float*    dinv  = (float*)   carve((size_t)N * 4);
  float*    stats = (float*)   carve(256 * 4);
  _Float16* W1t   = (_Float16*)carve((size_t)HID_DIM * IN_DIM * 2);
  _Float16* W2t   = (_Float16*)carve((size_t)OUT_DIM * HID_DIM * 2);
  _Float16* Ah    = (_Float16*)carve((size_t)N * IN_DIM * 2);
  _Float16* Hh    = (_Float16*)carve((size_t)N * HID_DIM * 2);
  float*    xw1   = (float*)   carve((size_t)N * HID_DIM * 4);
  float*    h     = (float*)   carve((size_t)N * HID_DIM * 4);
  float*    xw2   = xw1;   // layer-1 xw dead after edge_agg(1)
  float*    agg2  = h;     // h dead after f16 conversion

  const int TB = 256;
  auto blk = [](long long n, int tb) { return (int)((n + tb - 1) / tb); };

  // 1) symmetric normalization coefficients
  deg_init_kernel<<<blk(N, TB), TB, 0, stream>>>(dinv, N);
  deg_edges_kernel<<<blk(E, TB), TB, 0, stream>>>(ei, dinv, E);
  deg_to_dinv_kernel<<<blk(N, TB), TB, 0, stream>>>(dinv, N);

  // 2) f16 operands for WMMA
  cvt_f16_kernel<<<blk((long long)N * IN_DIM, TB), TB, 0, stream>>>(embed, Ah, (long long)N * IN_DIM);
  w_transpose_f16_kernel<<<blk(IN_DIM * HID_DIM, TB), TB, 0, stream>>>(W1, W1t, IN_DIM, HID_DIM);
  w_transpose_f16_kernel<<<blk(HID_DIM * OUT_DIM, TB), TB, 0, stream>>>(W2, W2t, HID_DIM, OUT_DIM);

  // 3) layer 1: GEMM -> init (self-loop + bias) -> edge scatter-add
  int waves1 = N / 16;
  wmma_gemm_f16_kernel<IN_DIM, HID_DIM>
      <<<blk((long long)waves1 * 32, TB), TB, 0, stream>>>(Ah, W1t, xw1, N);
  init_agg_kernel<<<blk((long long)N * HID_DIM, TB), TB, 0, stream>>>(xw1, dinv, b1, h, N, HID_DIM);
  edge_agg_kernel<<<blk((long long)E * (HID_DIM / 4), TB), TB, 0, stream>>>(ei, dinv, xw1, h, E, HID_DIM);

  // 4) layer 2
  cvt_f16_kernel<<<blk((long long)N * HID_DIM, TB), TB, 0, stream>>>(h, Hh, (long long)N * HID_DIM);
  wmma_gemm_f16_kernel<HID_DIM, OUT_DIM>
      <<<blk((long long)waves1 * 32, TB), TB, 0, stream>>>(Hh, W2t, xw2, N);
  init_agg_kernel<<<blk((long long)N * OUT_DIM, TB), TB, 0, stream>>>(xw2, dinv, b2, agg2, N, OUT_DIM);
  edge_agg_kernel<<<blk((long long)E * (OUT_DIM / 4), TB), TB, 0, stream>>>(ei, dinv, xw2, agg2, E, OUT_DIM);

  // 5) batch norm over nodes
  bn_zero_kernel<<<1, 128, 0, stream>>>(stats);
  bn_reduce_kernel<<<1024, 256, 0, stream>>>(agg2, stats, N);
  bn_finalize_kernel<<<1, 64, 0, stream>>>(stats, gamma, beta, N);
  bn_apply_kernel<<<blk((long long)N * OUT_DIM, TB), TB, 0, stream>>>(agg2, stats, out, N);
}
